// MainModel_5463198400972
// MI455X (gfx1250) — matrix operand order repaired and checked by reading, not compiled, not run on hardware
//
#include <hip/hip_runtime.h>

// =====================================================================
// MI455X (gfx1250) implementation of the BiLSTM sentence-pair model.
//
// Input flattening (setup_inputs dict order, lstm tuple flattened):
//  0 s1[32,256] i32   1 s2 i32
//  2 s1_h0[4,32,512]  3 s1_c0  4 s2_h0  5 s2_c0        (f32)
//  6 emb[50000,300]   (f32)
//  7..10  l0f: Wih[2048,300] Whh[2048,512] bih[2048] bhh[2048]
//  11..14 l0b, 15..18 l1f (Wih[2048,1024]), 19..22 l1b
//  23 ln_g[1024] 24 ln_b[1024]
//  25 mlp1_W[1024,2048] 26 mlp1_b[1024] 27 out_W[2,1024] 28 out_b[2]
// Output: [32,2] f32.
//
// Strategy:
//  * All big GEMMs in bf16 via v_wmma_f32_16x16x32_bf16 (f32 accumulate),
//    simple rolled k-loop (unroll 2) -> in-place accumulators, no phi movs.
//  * A-operand tensors written via swz32, the identity (lane half h holds k 16h..16h+15)
//    so each lane's A fragment is ONE contiguous 32B load.
//  * LSTM input projections batched over all timesteps; recurrence in a
//    persistent workgroup per direction (h in LDS bf16, c in registers,
//    Whh streamed from the 192MB L2).
//  * Attention-apply GEMM stages its [K,N] value panels into LDS with
//    global_load_async_to_lds_b128 (ASYNCcnt double-buffer pipeline).
// =====================================================================

typedef __bf16 bf16_t;
typedef __attribute__((ext_vector_type(16))) __bf16 v16bf;
typedef __attribute__((ext_vector_type(8)))  __bf16 v8bf;
typedef __attribute__((ext_vector_type(8)))  float  v8f;
typedef __attribute__((address_space(3))) char lds_char;

#define BB   32
#define LL   256
#define HH   512
#define DH   1024      // 2H
#define GG   2048      // 4H
#define EP   320       // padded embedding dim (300 -> 320)
#define ROWS 8192      // B*L

// ---------------------------------------------------------------- utils

__device__ __forceinline__ v8f zero_v8f() {
  v8f z;
#pragma unroll
  for (int i = 0; i < 8; ++i) z[i] = 0.0f;
  return z;
}

// A-operand column layout within each 32-col block: identity (contiguous k).
__device__ __forceinline__ int swz32(int c) {
  return c;
}

__device__ __forceinline__ float sigm_f(float x) {
  return 1.0f / (1.0f + __expf(-x));
}
__device__ __forceinline__ float tanh_f(float x) {
  x = fminf(fmaxf(x, -15.0f), 15.0f);
  float e = __expf(2.0f * x);
  return (e - 1.0f) / (e + 1.0f);
}

__device__ __forceinline__ float waveSum(float v) {
#pragma unroll
  for (int o = 16; o > 0; o >>= 1) v += __shfl_xor(v, o, 32);
  return v;
}
__device__ __forceinline__ float waveMax(float v) {
#pragma unroll
  for (int o = 16; o > 0; o >>= 1) v = fmaxf(v, __shfl_xor(v, o, 32));
  return v;
}
// blockDim.x must be a multiple of 32 and <= 256 for these.
__device__ __forceinline__ float blockSum(float v, float* red) {
  v = waveSum(v);
  if ((threadIdx.x & 31) == 0) red[threadIdx.x >> 5] = v;
  __syncthreads();
  int nw = blockDim.x >> 5;
  float r = 0.0f;
  for (int i = 0; i < nw; ++i) r += red[i];
  __syncthreads();
  return r;
}
__device__ __forceinline__ float blockMax(float v, float* red) {
  v = waveMax(v);
  if ((threadIdx.x & 31) == 0) red[threadIdx.x >> 5] = v;
  __syncthreads();
  int nw = blockDim.x >> 5;
  float r = red[0];
  for (int i = 1; i < nw; ++i) r = fmaxf(r, red[i]);
  __syncthreads();
  return r;
}

// ------------------------------------------------ WMMA tile load/store
// ISA layouts (CDNA5 ISA):
// A 16x32 bf16: lane l<16 -> row m0+l, elems a[0..7]=K[k0..k0+7],
//               a[8..15]=K[k0+16..k0+23]; lane l+16 -> K+8 / K+24.
// B 32x16 bf16: lane n<16 -> col n0+n, same k order as A; this code reads k0+16*(lane>=16)+i for A and B.
// C/D f32 16x16: c[r] = D[m0 + r + 8*(lane>=16)][n0 + (lane&15)].

// A from SWIZZLED storage: one contiguous 32B load per lane.
__device__ __forceinline__ v16bf load_a_swz(const bf16_t* X, int ldx,
                                            int m0, int k0, int lane) {
  int row = m0 + (lane & 15);
  int kb  = k0 + ((lane >> 4) << 4);
  return *(const v16bf*)(X + (size_t)row * ldx + kb);
}

// B source stored [N, K] row-major (weights / s2n in cos-GEMM): contiguous.
__device__ __forceinline__ v16bf load_b_nk(const bf16_t* W, int ldw,
                                           int n0, int k0, int lane) {
  int col = n0 + (lane & 15);
  int kb  = k0 + ((lane >> 4) << 4);
  return *(const v16bf*)(W + (size_t)col * ldw + kb);
}

__device__ __forceinline__ v8f wmma_bf16(v16bf a, v16bf b, v8f c) {
  return __builtin_amdgcn_wmma_f32_16x16x32_bf16(false, a, false, b,
                                                 (short)0, c, false, false);
}

// CDNA5 async copy 16B global -> LDS (ASYNCcnt-tracked, GV addressing).
__device__ __forceinline__ void async_g2l_b128(void* lds, const void* gptr) {
  asm volatile("global_load_async_to_lds_b128 %0, %1, off"
               :: "v"((lds_char*)lds), "v"(gptr)
               : "memory");
}
__device__ __forceinline__ void wait_asynccnt0() {
  asm volatile("s_wait_asynccnt 0" ::: "memory");
}

// ---------------------------------------------------------- small kernels

__global__ void cvt_pad_kernel(const float* __restrict__ src,
                               bf16_t* __restrict__ dst, int C, int Cp) {
  int r = blockIdx.x;
  for (int c = threadIdx.x; c < Cp; c += blockDim.x)
    dst[(size_t)r * Cp + c] = (c < C) ? (bf16_t)src[(size_t)r * C + c]
                                      : (bf16_t)0.0f;
}

__global__ void bias_sum_kernel(const float* __restrict__ a,
                                const float* __restrict__ b,
                                float* __restrict__ o, int n) {
  int i = blockIdx.x * blockDim.x + threadIdx.x;
  if (i < n) o[i] = a[i] + b[i];
}

// writes xb with A-swizzled columns
__global__ void embed_pad_kernel(const int* __restrict__ tok,
                                 const float* __restrict__ emb,
                                 bf16_t* __restrict__ xb) {
  int row = blockIdx.x;                       // b*L + t
  const float* e = emb + (size_t)tok[row] * 300;
  for (int c = threadIdx.x; c < EP; c += blockDim.x)
    xb[(size_t)row * EP + swz32(c)] =
        (c < 300) ? (bf16_t)e[c] : (bf16_t)0.0f;
}

// -------------------------------------------------------------- GEMM (W-style)
// C[M,N] = A[M,K](bf16, swizzled) @ W[N,K]^T(bf16) + bias, optional ReLU.
__global__ void __launch_bounds__(256)
gemm_bf16_nk(const bf16_t* __restrict__ A, int lda,
             const bf16_t* __restrict__ W, int ldw,
             const float* __restrict__ bias,
             float* __restrict__ C, int ldc,
             int M, int N, int K, int act) {
  int gw   = (blockIdx.x * blockDim.x + threadIdx.x) >> 5;
  int lane = threadIdx.x & 31;
  int tn   = N >> 6;                          // 64-col tiles
  int tm   = M >> 4;
  if (gw >= tm * tn) return;
  int m0 = (gw / tn) << 4;
  int n0 = (gw % tn) << 6;

  v8f acc[4];
#pragma unroll
  for (int j = 0; j < 4; ++j) acc[j] = zero_v8f();

#pragma unroll 2
  for (int k0 = 0; k0 < K; k0 += 32) {
    v16bf a = load_a_swz(A, lda, m0, k0, lane);
    v16bf bt[4];
#pragma unroll
    for (int j = 0; j < 4; ++j)
      bt[j] = load_b_nk(W, ldw, n0 + j * 16, k0, lane);
#pragma unroll
    for (int j = 0; j < 4; ++j) acc[j] = wmma_bf16(a, bt[j], acc[j]);
  }

#pragma unroll
  for (int j = 0; j < 4; ++j) {
    int col = n0 + j * 16 + (lane & 15);
    float bv = bias ? bias[col] : 0.0f;
    int rb = m0 + ((lane >> 4) << 3);
#pragma unroll
    for (int r = 0; r < 8; ++r) {
      float v = acc[j][r] + bv;
      if (act == 1) v = fmaxf(v, 0.0f);
      C[(size_t)(rb + r) * ldc + col] = v;
    }
  }
}

// ---------------------------------------------------------------- LSTM step
// One block per direction (blockIdx.x: 0=fwd, 1=bwd); 1024 threads = 32 waves.
// Wave w owns h columns [w*16, w*16+16); computes gate tiles i,f,g,o for
// both batch halves. gin holds precomputed x@Wih^T + (bih+bhh), rows b*L+t.
// h is kept A-swizzled in LDS; out_b16 is written A-swizzled (feeds next
// layer's input GEMM), out_f32 linear (feeds LayerNorm).
__global__ void __launch_bounds__(1024)
lstm_layer_kernel(const float* __restrict__ gin_f,
                  const float* __restrict__ gin_b,
                  const bf16_t* __restrict__ Whh_f,
                  const bf16_t* __restrict__ Whh_b,
                  const float* __restrict__ h0,   // layer base [2,32,512]
                  const float* __restrict__ c0,
                  bf16_t* __restrict__ out_b16,   // [8192,1024] swizzled
                  float*  __restrict__ out_f32) { // [8192,1024] linear
  __shared__ bf16_t h_lds[BB][HH];               // 32 KB
  const int dir = blockIdx.x;
  const float*  gin = dir ? gin_b : gin_f;
  const bf16_t* Whh = dir ? Whh_b : Whh_f;
  const float*  h0d = h0 + (size_t)dir * BB * HH;
  const float*  c0d = c0 + (size_t)dir * BB * HH;
  const int ocol0 = dir * HH;

  const int tid  = threadIdx.x;
  const int wave = tid >> 5;
  const int lane = tid & 31;
  const int n0   = wave << 4;                    // h column block
  const int col  = n0 + (lane & 15);             // this lane's h column
  const int cols = swz32(col);                   // swizzled position

  // init h (LDS, bf16, swizzled)
  for (int idx = tid; idx < BB * HH; idx += blockDim.x) {
    int bi = idx >> 9, ci = idx & (HH - 1);
    h_lds[bi][swz32(ci)] = (bf16_t)h0d[idx];
  }

  // init c (registers, C/D layout)
  float cst[2][8];
#pragma unroll
  for (int half = 0; half < 2; ++half) {
    int rb = half * 16 + ((lane >> 4) << 3);
#pragma unroll
    for (int r = 0; r < 8; ++r)
      cst[half][r] = c0d[(size_t)(rb + r) * HH + col];
  }
  __syncthreads();

  for (int step = 0; step < LL; ++step) {
    const int t = dir ? (LL - 1 - step) : step;

    // speculative prefetch of next step's gate inputs (L2 warm-up)
    {
      int tn_ = dir ? (t > 0 ? t - 1 : 0) : (t < LL - 1 ? t + 1 : t);
      __builtin_prefetch(
          &gin[((size_t)((lane >> 4) << 3) * LL + tn_) * GG + n0 + (lane & 15)],
          0, 0);
    }

    v8f acc[4][2];
#pragma unroll
    for (int g = 0; g < 4; ++g) {
      int gcol = n0 + g * HH + (lane & 15);
#pragma unroll
      for (int half = 0; half < 2; ++half) {
        int rb = half * 16 + ((lane >> 4) << 3);
        v8f a;
#pragma unroll
        for (int r = 0; r < 8; ++r)
          a[r] = gin[((size_t)(rb + r) * LL + t) * GG + gcol];
        acc[g][half] = a;
      }
    }

    for (int k0 = 0; k0 < HH; k0 += 32) {
      v16bf a0 = load_a_swz(&h_lds[0][0], HH, 0,  k0, lane);
      v16bf a1 = load_a_swz(&h_lds[0][0], HH, 16, k0, lane);
      v16bf bg[4];
#pragma unroll
      for (int g = 0; g < 4; ++g)
        bg[g] = load_b_nk(Whh, HH, n0 + g * HH, k0, lane);
#pragma unroll
      for (int g = 0; g < 4; ++g) {
        acc[g][0] = wmma_bf16(a0, bg[g], acc[g][0]);
        acc[g][1] = wmma_bf16(a1, bg[g], acc[g][1]);
      }
    }
    __syncthreads();  // everyone done reading h_lds for this step

#pragma unroll
    for (int half = 0; half < 2; ++half) {
      int rb = half * 16 + ((lane >> 4) << 3);
#pragma unroll
      for (int r = 0; r < 8; ++r) {
        float iv = sigm_f(acc[0][half][r]);
        float fv = sigm_f(acc[1][half][r]);
        float gv = tanh_f(acc[2][half][r]);
        float ov = sigm_f(acc[3][half][r]);
        float c  = fv * cst[half][r] + iv * gv;
        cst[half][r] = c;
        float h  = ov * tanh_f(c);
        int bi = rb + r;
        h_lds[bi][cols] = (bf16_t)h;
        size_t obase = ((size_t)bi * LL + t) * DH;
        out_f32[obase + ocol0 + col] = h;
        out_b16[obase + ocol0 + cols] = (bf16_t)h;  // swizzled (ocol0%32==0)
      }
    }
    __syncthreads();  // h_lds updated for next step
  }
}

// ----------------------------------------------------------- LayerNorm (D=1024)
// Outputs: Yf linear f32, Yb linear bf16 (B-operand use), Ya A-swizzled bf16
// (A-operand use), norms (post-LN row L2 norm).
__global__ void ln_kernel(const float* __restrict__ X,
                          const float* __restrict__ g,
                          const float* __restrict__ bb,
                          float* __restrict__ Yf,
                          bf16_t* __restrict__ Yb,
                          bf16_t* __restrict__ Ya,
                          float* __restrict__ norms, int aux) {
  __shared__ float red[8];
  const int D = DH;
  int row = blockIdx.x;
  const float* x = X + (size_t)row * D;
  float s = 0.0f;
  for (int i = threadIdx.x; i < D; i += blockDim.x) s += x[i];
  float mu = blockSum(s, red) * (1.0f / D);
  float vs = 0.0f;
  for (int i = threadIdx.x; i < D; i += blockDim.x) {
    float d = x[i] - mu; vs += d * d;
  }
  float var = blockSum(vs, red) * (1.0f / D);
  float inv = rsqrtf(var + 1e-5f);
  float sq = 0.0f;
  for (int i = threadIdx.x; i < D; i += blockDim.x) {
    float y = (x[i] - mu) * inv * g[i] + bb[i];
    Yf[(size_t)row * D + i] = y;
    if (aux) {
      Yb[(size_t)row * D + i] = (bf16_t)y;
      Ya[(size_t)row * D + swz32(i)] = (bf16_t)y;
    }
    sq += y * y;
  }
  if (aux) {
    float nr = blockSum(sq, red);
    if (threadIdx.x == 0) norms[row] = sqrtf(nr);
  }
}

// ------------------------------------------------- cosine co-attention dots
// cos[b,i,j] = <s1n[b,i,:], s2n[b,j,:]> / max(n1*n2, 1e-8). K = 1024.
// S1a is A-swizzled bf16; S2 is linear bf16 (also a B operand elsewhere).
__global__ void __launch_bounds__(256)
gemm_cos(const bf16_t* __restrict__ S1a,
         const bf16_t* __restrict__ S2,
         const float* __restrict__ n1,
         const float* __restrict__ n2,
         float* __restrict__ Cos) {
  int gw   = (blockIdx.x * blockDim.x + threadIdx.x) >> 5;
  int lane = threadIdx.x & 31;
  int b  = gw >> 6;                    // 64 waves per batch (16 mt x 4 nt)
  if (b >= BB) return;
  int w  = gw & 63;
  int m0 = (w >> 2) << 4;
  int n0 = (w & 3) << 6;
  const bf16_t* A  = S1a + (size_t)b * LL * DH;
  const bf16_t* Bw = S2  + (size_t)b * LL * DH;

  v8f acc[4];
#pragma unroll
  for (int j = 0; j < 4; ++j) acc[j] = zero_v8f();

#pragma unroll 2
  for (int k0 = 0; k0 < DH; k0 += 32) {
    v16bf a = load_a_swz(A, DH, m0, k0, lane);
    v16bf bt[4];
#pragma unroll
    for (int j = 0; j < 4; ++j)
      bt[j] = load_b_nk(Bw, DH, n0 + j * 16, k0, lane);
#pragma unroll
    for (int j = 0; j < 4; ++j) acc[j] = wmma_bf16(a, bt[j], acc[j]);
  }

  const float* N1 = n1 + b * LL;
  const float* N2 = n2 + b * LL;
#pragma unroll
  for (int j = 0; j < 4; ++j) {
    int colj = n0 + j * 16 + (lane & 15);
    float nv2 = N2[colj];
    int rb = m0 + ((lane >> 4) << 3);
#pragma unroll
    for (int r = 0; r < 8; ++r) {
      float den = fmaxf(N1[rb + r] * nv2, 1e-8f);
      Cos[((size_t)b * LL + rb + r) * LL + colj] = acc[j][r] / den;
    }
  }
}

// ----------------------------------------------------------------- softmax
// P (attention weights) is consumed only as an A operand -> store swizzled.
__global__ void softmax_rows(const float* __restrict__ Cc,
                             bf16_t* __restrict__ P) {
  __shared__ float red[8];
  int row = blockIdx.x;                        // b*L + i
  float v = Cc[(size_t)row * LL + threadIdx.x];
  float m = blockMax(v, red);
  float e = __expf(v - m);
  float s = blockSum(e, red);
  P[(size_t)row * LL + swz32(threadIdx.x)] = (bf16_t)(e / s);
}

// P2T[b,j,i] = softmax over i of cos[b,i,j]  (i = K dim -> swizzled)
__global__ void softmax_cols(const float* __restrict__ Cc,
                             bf16_t* __restrict__ P2T) {
  __shared__ float red[8];
  int b = blockIdx.x >> 8;
  int j = blockIdx.x & 255;
  float v = Cc[((size_t)b * LL + threadIdx.x) * LL + j];
  float m = blockMax(v, red);
  float e = __expf(v - m);
  float s = blockSum(e, red);
  P2T[((size_t)b * LL + j) * LL + swz32(threadIdx.x)] = (bf16_t)(e / s);
}

// ------------------------------------------------- attention apply (per-batch)
// O[b,m,d] = sum_k P[b,m,k] * Vv[b,k,d];  P:[32,256,256] swizzled A,
// Vv:[32,256,1024] linear. Block = 8 waves sharing one 64-wide n-panel; the
// [32 x 64] value panel per k-step is staged into LDS with
// global_load_async_to_lds_b128 and double-buffered on ASYNCcnt.
#define ATT_PAD 72   // row stride (elems); 144B keeps 16B-aligned b128 chunks
__global__ void __launch_bounds__(256)
gemm_att(const bf16_t* __restrict__ P,
         const bf16_t* __restrict__ Vv,
         float* __restrict__ O) {
  __shared__ bf16_t bpanel[2][32][ATT_PAD];
  const int blk  = blockIdx.x;        // b(32) x mblk(2) x npanel(16) = 1024
  const int b    = blk >> 5;
  const int mblk = (blk >> 4) & 1;
  const int np   = blk & 15;
  const int tid  = threadIdx.x;
  const int wave = tid >> 5;
  const int lane = tid & 31;
  const int m0   = mblk * 128 + wave * 16;
  const int n0   = np * 64;
  const bf16_t* A  = P  + (size_t)b * LL * LL;
  const bf16_t* Vb = Vv + (size_t)b * LL * DH;

  // each thread moves one 16B chunk of the 32x64 panel per stage
  const int krow = tid >> 3;          // 0..31
  const int nch  = (tid & 7) * 8;     // bf16 chunk start within panel row

  {
    const bf16_t* g0 = Vb + (size_t)krow * DH + n0 + nch;
    async_g2l_b128(&bpanel[0][krow][nch], g0);
  }
  wait_asynccnt0();
  __syncthreads();

  v8f acc[4];
#pragma unroll
  for (int j = 0; j < 4; ++j) acc[j] = zero_v8f();

  for (int kk = 0; kk < 8; ++kk) {     // K = 256 = 8 * 32
    int cur = kk & 1;
    if (kk + 1 < 8) {
      const bf16_t* g =
          Vb + (size_t)((kk + 1) * 32 + krow) * DH + n0 + nch;
      async_g2l_b128(&bpanel[cur ^ 1][krow][nch], g);
    }
    v16bf a = load_a_swz(A, LL, m0, kk * 32, lane);
#pragma unroll
    for (int j = 0; j < 4; ++j) {
      int colc = j * 16 + (lane & 15);
      int kb   = (lane >> 4) << 4;
      v16bf bt;
#pragma unroll
      for (int i = 0; i < 16; ++i) bt[i] = bpanel[cur][kb + i][colc];
      acc[j] = wmma_bf16(a, bt, acc[j]);
    }
    wait_asynccnt0();
    __syncthreads();
  }

#pragma unroll
  for (int j = 0; j < 4; ++j) {
    int colj = n0 + j * 16 + (lane & 15);
    int rb = m0 + ((lane >> 4) << 3);
#pragma unroll
    for (int r = 0; r < 8; ++r)
      O[((size_t)b * LL + rb + r) * DH + colj] = acc[j][r];
  }
}

// -------------------------------------------------------------- pooling / MLP
__global__ void meanpool_kernel(const float* __restrict__ X,
                                float* __restrict__ R) {
  int b = blockIdx.x, d = threadIdx.x;           // blockDim = 1024
  float s = 0.0f;
  for (int i = 0; i < LL; ++i) s += X[((size_t)b * LL + i) * DH + d];
  R[(size_t)b * DH + d] = s * (1.0f / LL);
}

__global__ void merged_kernel(const float* __restrict__ r1,
                              const float* __restrict__ r2,
                              float* __restrict__ mf,
                              bf16_t* __restrict__ mb) {
  int b = blockIdx.x;
  for (int d = threadIdx.x; d < GG; d += blockDim.x) {
    float v;
    if (d < DH) v = r1[b * DH + d] + r2[b * DH + d];
    else { float df = r1[b * DH + d - DH] - r2[b * DH + d - DH]; v = df * df; }
    mf[(size_t)b * GG + d] = v;
    mb[(size_t)b * GG + swz32(d)] = (bf16_t)v;   // A-swizzled for MLP GEMM
  }
}

__global__ void final_out_kernel(const float* __restrict__ hid,
                                 const float* __restrict__ oW,
                                 const float* __restrict__ ob,
                                 float* __restrict__ out) {
  int t = threadIdx.x;
  if (t >= 64) return;
  int b = t >> 1, c = t & 1;
  float s = ob[c];
  for (int k = 0; k < DH; ++k) s += hid[(size_t)b * DH + k] * oW[c * DH + k];
  out[b * 2 + c] = 1.0f / (1.0f + __expf(-s));
}

// =====================================================================
extern "C" void kernel_launch(void* const* d_in, const int* in_sizes, int n_in,
                              void* d_out, int out_size, void* d_ws,
                              size_t ws_size, hipStream_t stream) {
  (void)in_sizes; (void)n_in; (void)out_size; (void)ws_size;
  const int*   s1    = (const int*)d_in[0];
  const int*   s2    = (const int*)d_in[1];
  const float* h0s[2] = {(const float*)d_in[2], (const float*)d_in[4]};
  const float* c0s[2] = {(const float*)d_in[3], (const float*)d_in[5]};
  const float* emb   = (const float*)d_in[6];
  const float *Wih[4], *Whh[4], *bih[4], *bhh[4];
  for (int i = 0; i < 4; ++i) {
    Wih[i] = (const float*)d_in[7 + 4 * i + 0];
    Whh[i] = (const float*)d_in[7 + 4 * i + 1];
    bih[i] = (const float*)d_in[7 + 4 * i + 2];
    bhh[i] = (const float*)d_in[7 + 4 * i + 3];
  }
  const float* ln_g   = (const float*)d_in[23];
  const float* ln_b   = (const float*)d_in[24];
  const float* mlp1_W = (const float*)d_in[25];
  const float* mlp1_b = (const float*)d_in[26];
  const float* out_W  = (const float*)d_in[27];
  const float* out_b  = (const float*)d_in[28];
  float* dout = (float*)d_out;

  // ---------------- workspace carve-up (assumes ws_size >= ~380 MB)
  char* base = (char*)d_ws;
  size_t off = 0;
  auto alloc = [&](size_t bytes) -> void* {
    off = (off + 255) & ~(size_t)255;
    void* p = base + off;
    off += bytes;
    return p;
  };
  bf16_t* Wih0b[2]; bf16_t* Whhb[4]; bf16_t* Wih1b[2];
  Wih0b[0] = (bf16_t*)alloc((size_t)GG * EP * 2);
  Wih0b[1] = (bf16_t*)alloc((size_t)GG * EP * 2);
  for (int i = 0; i < 4; ++i) Whhb[i] = (bf16_t*)alloc((size_t)GG * HH * 2);
  Wih1b[0] = (bf16_t*)alloc((size_t)GG * DH * 2);
  Wih1b[1] = (bf16_t*)alloc((size_t)GG * DH * 2);
  bf16_t* mlp1b16 = (bf16_t*)alloc((size_t)DH * GG * 2);
  float*  bsum    = (float*)alloc((size_t)4 * GG * 4);
  bf16_t* xb      = (bf16_t*)alloc((size_t)ROWS * EP * 2);
  float*  gin_f   = (float*)alloc((size_t)ROWS * GG * 4);
  float*  gin_b   = (float*)alloc((size_t)ROWS * GG * 4);
  bf16_t* out_l0  = (bf16_t*)alloc((size_t)ROWS * DH * 2);
  float*  out_raw = (float*)alloc((size_t)ROWS * DH * 4);
  float*  sn_f[2]; bf16_t* sn_b[2]; bf16_t* sn_a[2]; float* norms[2];
  for (int i = 0; i < 2; ++i) {
    sn_f[i]  = (float*)alloc((size_t)ROWS * DH * 4);
    sn_b[i]  = (bf16_t*)alloc((size_t)ROWS * DH * 2);
    sn_a[i]  = (bf16_t*)alloc((size_t)ROWS * DH * 2);
    norms[i] = (float*)alloc((size_t)ROWS * 4);
  }
  float*  cosb = (float*)alloc((size_t)BB * LL * LL * 4);
  bf16_t* P1   = (bf16_t*)alloc((size_t)BB * LL * LL * 2);
  bf16_t* P2T  = (bf16_t*)alloc((size_t)BB * LL * LL * 2);
  float*  r1   = (float*)alloc((size_t)BB * DH * 4);
  float*  r2   = (float*)alloc((size_t)BB * DH * 4);
  float*  mgf  = (float*)alloc((size_t)BB * GG * 4);
  bf16_t* mgb  = (bf16_t*)alloc((size_t)BB * GG * 2);
  float*  hid  = (float*)alloc((size_t)BB * DH * 4);
  // overlays: gin buffers are dead once both sentences are encoded
  float* att1_raw = gin_f;
  float* att2_raw = gin_f + (size_t)ROWS * DH;
  float* att1n    = gin_b;
  float* att2n    = gin_b + (size_t)ROWS * DH;

  auto gemm = [&](const bf16_t* A, int lda, const bf16_t* W, int ldw,
                  const float* bias, float* C, int ldc,
                  int M, int N, int K, int act) {
    int waves = (M >> 4) * (N >> 6);
    int blocks = (waves * 32 + 255) / 256;
    gemm_bf16_nk<<<blocks, 256, 0, stream>>>(A, lda, W, ldw, bias, C, ldc,
                                             M, N, K, act);
  };

  // ---- weight conversion (f32 -> bf16, pad K of layer-0 Wih to 320)
  cvt_pad_kernel<<<GG, 256, 0, stream>>>(Wih[0], Wih0b[0], 300, EP);
  cvt_pad_kernel<<<GG, 256, 0, stream>>>(Wih[1], Wih0b[1], 300, EP);
  for (int i = 0; i < 4; ++i)
    cvt_pad_kernel<<<GG, 256, 0, stream>>>(Whh[i], Whhb[i], HH, HH);
  cvt_pad_kernel<<<GG, 256, 0, stream>>>(Wih[2], Wih1b[0], DH, DH);
  cvt_pad_kernel<<<GG, 256, 0, stream>>>(Wih[3], Wih1b[1], DH, DH);
  cvt_pad_kernel<<<DH, 256, 0, stream>>>(mlp1_W, mlp1b16, GG, GG);
  for (int i = 0; i < 4; ++i)
    bias_sum_kernel<<<8, 256, 0, stream>>>(bih[i], bhh[i], bsum + i * GG, GG);

  // ---- encode both sentences
  for (int s = 0; s < 2; ++s) {
    const int* tok = s ? s2 : s1;
    embed_pad_kernel<<<ROWS, 320, 0, stream>>>(tok, emb, xb);
    // layer 0 input projections (K padded to 320; pad region is zero)
    gemm(xb, EP, Wih0b[0], EP, bsum + 0 * GG, gin_f, GG, ROWS, GG, EP, 0);
    gemm(xb, EP, Wih0b[1], EP, bsum + 1 * GG, gin_b, GG, ROWS, GG, EP, 0);
    lstm_layer_kernel<<<2, 1024, 0, stream>>>(gin_f, gin_b, Whhb[0], Whhb[1],
                                              h0s[s], c0s[s], out_l0, out_raw);
    // layer 1 input projections (input = concat(hf,hb), K = 1024)
    gemm(out_l0, DH, Wih1b[0], DH, bsum + 2 * GG, gin_f, GG, ROWS, GG, DH, 0);
    gemm(out_l0, DH, Wih1b[1], DH, bsum + 3 * GG, gin_b, GG, ROWS, GG, DH, 0);
    lstm_layer_kernel<<<2, 1024, 0, stream>>>(
        gin_f, gin_b, Whhb[2], Whhb[3],
        h0s[s] + 2 * BB * HH, c0s[s] + 2 * BB * HH, out_l0, out_raw);
    ln_kernel<<<ROWS, 256, 0, stream>>>(out_raw, ln_g, ln_b, sn_f[s], sn_b[s],
                                        sn_a[s], norms[s], 1);
  }

  // ---- co-attention
  gemm_cos<<<(BB * 64 * 32) / 256, 256, 0, stream>>>(sn_a[0], sn_b[1],
                                                     norms[0], norms[1], cosb);
  softmax_rows<<<ROWS, 256, 0, stream>>>(cosb, P1);
  softmax_cols<<<ROWS, 256, 0, stream>>>(cosb, P2T);
  gemm_att<<<(BB * 256 * 32) / 256, 256, 0, stream>>>(P1,  sn_b[1], att1_raw);
  gemm_att<<<(BB * 256 * 32) / 256, 256, 0, stream>>>(P2T, sn_b[0], att2_raw);
  ln_kernel<<<ROWS, 256, 0, stream>>>(att1_raw, ln_g, ln_b, att1n, nullptr,
                                      nullptr, nullptr, 0);
  ln_kernel<<<ROWS, 256, 0, stream>>>(att2_raw, ln_g, ln_b, att2n, nullptr,
                                      nullptr, nullptr, 0);

  // ---- pooling + MLP head
  meanpool_kernel<<<BB, 1024, 0, stream>>>(att1n, r1);
  meanpool_kernel<<<BB, 1024, 0, stream>>>(att2n, r2);
  merged_kernel<<<BB, 1024, 0, stream>>>(r1, r2, mgf, mgb);
  gemm(mgb, GG, mlp1b16, GG, mlp1_b, hid, DH, BB, DH, GG, 1);
  final_out_kernel<<<1, 64, 0, stream>>>(hid, out_W, out_b, dout);
}
